// MorphoGradDGNN_10393820856803
// MI455X (gfx1250) — matrix operand order repaired and checked by reading, not compiled, not run on hardware
//
#include <hip/hip_runtime.h>
#include <hip/hip_bf16.h>

typedef __attribute__((ext_vector_type(16))) _Float16 v16h;
typedef __attribute__((ext_vector_type(8)))  _Float16 v8h;
typedef __attribute__((ext_vector_type(8)))  float    v8f;

#define N_PTS 8192
#define KNN   20
#define CH    64

// ---------------------------------------------------------------- WMMA helper
__device__ __forceinline__ v8f wmma_f16f32(v16h a, v16h b, v8f c) {
  // D = A(16x32 f16) * B(32x16 f16) + C(16x16 f32)
  return __builtin_amdgcn_wmma_f32_16x16x32_f16(
      /*neg_a=*/false, a, /*neg_b=*/false, b,
      /*c_mod=*/(short)0, c, /*reuse_a=*/false, /*reuse_b=*/false);
}

// --------------------------------------------- async global->LDS (CDNA5 DMA)
// GLOBAL_LOAD_ASYNC_TO_LDS_B128, GVS mode: SGPR64 base + per-lane u32 offset,
// per-lane LDS byte address in VDST. Tracked by ASYNCcnt (in-order complete).
__device__ __forceinline__ void async_ld_b128(void* lds_dst, const void* base,
                                              unsigned byte_off) {
  asm volatile("global_load_async_to_lds_b128 %0, %1, %2"
               :: "v"((unsigned)(uintptr_t)lds_dst),   // LDS byte address
                  "v"(byte_off),                       // per-lane byte offset
                  "s"((unsigned long long)(uintptr_t)base)
               : "memory");
}
__device__ __forceinline__ void wait_async_le2() {
#if __has_builtin(__builtin_amdgcn_s_wait_asynccnt)
  __builtin_amdgcn_s_wait_asynccnt(2);
#else
  asm volatile("s_wait_asynccnt 2" ::: "memory");
#endif
}
__device__ __forceinline__ void wait_async_le0() {
#if __has_builtin(__builtin_amdgcn_s_wait_asynccnt)
  __builtin_amdgcn_s_wait_asynccnt(0);
#else
  asm volatile("s_wait_asynccnt 0" ::: "memory");
#endif
}

// Running top-K (ascending) insertion in LDS; common case is one compare.
__device__ __forceinline__ void topk_insert(float* td, int* ti, float d, int id) {
  if (d >= td[KNN - 1]) return;
  int j = KNN - 1;
  while (j > 0 && td[j - 1] > d) { td[j] = td[j - 1]; ti[j] = ti[j - 1]; --j; }
  td[j] = d; ti[j] = id;
}

// ---------------------------------------------------------------- kNN in 3-D
__global__ void knn3_kernel(const float* __restrict__ x, int* __restrict__ idx) {
  __shared__ float cxyz[128 * 3];
  __shared__ float topd[128 * KNN];
  __shared__ int   topi[128 * KNN];
  const int t = threadIdx.x;                 // one query per thread
  const int q = blockIdx.x * 128 + t;
  const float qx = x[q * 3 + 0], qy = x[q * 3 + 1], qz = x[q * 3 + 2];
  float* td = &topd[t * KNN]; int* ti = &topi[t * KNN];
  for (int j = 0; j < KNN; ++j) { td[j] = __builtin_inff(); ti[j] = 0; }
  for (int cb = 0; cb < N_PTS; cb += 128) {
    __syncthreads();
    for (int i = t; i < 128 * 3; i += 128) cxyz[i] = x[cb * 3 + i];
    if (cb + 128 < N_PTS) __builtin_prefetch(&x[(cb + 128) * 3], 0, 1);
    __syncthreads();
    for (int j = 0; j < 128; ++j) {
      const int cid = cb + j;
      const float dx = cxyz[j * 3 + 0] - qx;
      const float dy = cxyz[j * 3 + 1] - qy;
      const float dz = cxyz[j * 3 + 2] - qz;
      const float d = dx * dx + dy * dy + dz * dz;
      if (cid == q) continue;
      topk_insert(td, ti, d, cid);
    }
  }
  for (int j = 0; j < KNN; ++j) idx[q * KNN + j] = ti[j];
}

// ------------------------------------------------------- EdgeConv (first layer)
__global__ void edgeconv_kernel(const float* __restrict__ x, const int* __restrict__ idx,
                                const float* __restrict__ W1, const float* __restrict__ b1,
                                float* __restrict__ xfeat) {
  __shared__ float nb[KNN * 3];
  __shared__ float q[3];
  const int n = blockIdx.x, c = threadIdx.x;   // 64 threads = out channels
  if (c < 3) q[c] = x[n * 3 + c];
  if (c < KNN) {
    const int p = idx[n * KNN + c];
    nb[c * 3 + 0] = x[p * 3 + 0];
    nb[c * 3 + 1] = x[p * 3 + 1];
    nb[c * 3 + 2] = x[p * 3 + 2];
  }
  __syncthreads();
  const float w0 = W1[0 * CH + c], w1 = W1[1 * CH + c], w2 = W1[2 * CH + c];
  const float w3 = W1[3 * CH + c], w4 = W1[4 * CH + c], w5 = W1[5 * CH + c];
  const float f0 = q[0], f1 = q[1], f2 = q[2];
  const float base = b1[c] + f0 * w0 + f1 * w1 + f2 * w2;
  float best = 0.f;
  for (int j = 0; j < KNN; ++j) {
    const float h = base + (nb[j * 3 + 0] - f0) * w3
                         + (nb[j * 3 + 1] - f1) * w4
                         + (nb[j * 3 + 2] - f2) * w5;
    best = fmaxf(best, fmaxf(h, 0.f));
  }
  xfeat[n * CH + c] = best;
}

// ------------------------------------- squared norms + f16 copy of feature rows
__global__ void sqnorm_kernel(const float* __restrict__ X, int stride,
                              _Float16* __restrict__ F16, float* __restrict__ SQ) {
  __shared__ float red[CH];
  const int n = blockIdx.x, c = threadIdx.x;
  const float v = X[n * stride + c];
  F16[n * CH + c] = (_Float16)v;
  red[c] = v * v;
  __syncthreads();
  for (int s = 32; s > 0; s >>= 1) {
    if (c < s) red[c] += red[c + s];
    __syncthreads();
  }
  if (c == 0) SQ[n] = red[0];
}

// ----------------------------------------------- fused WMMA distances + top-K
#define KQT 128   // queries per workgroup (1 selection thread each)
#define KCT 32    // candidate chunk
__global__ void knn64_kernel(const _Float16* __restrict__ F16,
                             const float* __restrict__ SQ,
                             int* __restrict__ idx) {
  __shared__ __align__(16) _Float16 qf[KQT * CH];      // 16 KB query tile
  __shared__ __align__(16) _Float16 cf[2][KCT * CH];   // 2 x 4 KB cand tiles
  __shared__ float dist[KQT * KCT];                    // 16 KB distances
  __shared__ float sqQ[KQT];
  __shared__ float sqC[2][KCT];
  __shared__ float topd[KQT * KNN];
  __shared__ int   topi[KQT * KNN];
  const int t = threadIdx.x;           // 0..127 (4 waves)
  const int lane = t & 31, wave = t >> 5;
  const int qbase = blockIdx.x * KQT;

  { // query tile: each thread copies its own 128B row
    const uint4* src = (const uint4*)&F16[(qbase + t) * CH];
    uint4* dst = (uint4*)&qf[t * CH];
#pragma unroll
    for (int i = 0; i < 8; ++i) dst[i] = src[i];
    sqQ[t] = SQ[qbase + t];
  }
  float* td = &topd[t * KNN]; int* ti = &topi[t * KNN];
  for (int j = 0; j < KNN; ++j) { td[j] = __builtin_inff(); ti[j] = 0; }

  // async-stage a candidate chunk (KCT rows * 128B = 256 x b128; 2 per thread)
  auto stage = [&](int cb, int buf) {
    const unsigned cbase = (unsigned)cb * (CH * 2);
    uint4* d = (uint4*)cf[buf];
    async_ld_b128(&d[t],       F16, cbase + (unsigned)t * 16u);
    async_ld_b128(&d[t + 128], F16, cbase + (unsigned)(t + 128) * 16u);
  };
  stage(0, 0);

  for (int cb = 0; cb < N_PTS; cb += KCT) {
    const int buf = (cb / KCT) & 1;
    if (cb + KCT < N_PTS) { stage(cb + KCT, buf ^ 1); wait_async_le2(); }
    else                  { wait_async_le0(); }
    if (t < KCT) sqC[buf][t] = SQ[cb + t];
    __syncthreads();   // barrier A: cf[buf]/sqC[buf] visible; dist reusable

    const _Float16* cfb = cf[buf];
    // 8 row-tiles x 2 col-tiles of 16x16; constant trip count -> no EXEC masking
#pragma unroll
    for (int tt = 0; tt < 4; ++tt) {
      const int T = (tt << 2) | wave;
      const int mi = T >> 1, ni = T & 1;
      v8f acc = {};
#pragma unroll
      for (int ks = 0; ks < 2; ++ks) {   // K = 64 in two 32-steps
        // A fragment (16x32 f16): 16 contiguous K at (lane>>4)*16
        const int arow = (mi << 4) + (lane & 15);
        const int ab   = ((lane >> 4) << 4) + (ks << 5);
        v16h a = *(const v16h*)&qf[arow * CH + ab];
        // B fragment (32x16 f16): N = lane&15, 16 contiguous K at (lane>>4)*16
        const int bcol = (ni << 4) + (lane & 15);
        const int bb   = ((lane >> 4) << 4) + (ks << 5);
        v16h b = *(const v16h*)&cfb[bcol * CH + bb];
        acc = wmma_f16f32(a, b, acc);
      }
      // C/D layout: VGPR v -> M = v + (lane>>4)*8, N = lane&15
      const int nn = (ni << 4) + (lane & 15);
      const int mb = (mi << 4) + ((lane >> 4) << 3);
      const float sc = sqC[buf][nn];
#pragma unroll
      for (int v = 0; v < 8; ++v) {
        const int mm = mb + v;
        float d = sqQ[mm] + sc - 2.f * acc[v];
        if (qbase + mm == cb + nn) d = __builtin_inff();   // exclude self
        dist[mm * KCT + nn] = d;
      }
    }
    __syncthreads();   // barrier B: dist complete
#pragma unroll 4
    for (int j = 0; j < KCT; ++j)
      topk_insert(td, ti, dist[t * KCT + j], cb + j);
  }
  __syncthreads();
  for (int j = 0; j < KNN; ++j) idx[(qbase + t) * KNN + j] = ti[j];
}

// ------------------- dilation (max over kNN) + MLP + ReLU - input, f32+f16 out
__global__ void dilate_kernel(const float* __restrict__ feat, int inStride,
                              const int* __restrict__ idx,
                              const float* __restrict__ Wd, const float* __restrict__ bd,
                              float* __restrict__ out, _Float16* __restrict__ out16,
                              int outStride) {
  __shared__ float dil[CH];
  __shared__ int nb[KNN];
  const int n = blockIdx.x, c = threadIdx.x;
  if (c < KNN) nb[c] = idx[n * KNN + c];
  __syncthreads();
  float m = -__builtin_inff();
  for (int j = 0; j < KNN; ++j) m = fmaxf(m, feat[nb[j] * inStride + c]);
  dil[c] = m;
  __syncthreads();
  float s = bd[c];
  for (int k = 0; k < CH; ++k) s = fmaf(dil[k], Wd[k * CH + c], s);
  const float r = fmaxf(s, 0.f) - feat[n * inStride + c];
  out[n * outStride + c]   = r;
  out16[n * outStride + c] = (_Float16)r;
}

// ----------------------- f32 [K,N] -> f16 [Npad,K] transpose (zero-pad rows)
__global__ void cvtT_kernel(const float* __restrict__ src, _Float16* __restrict__ dst,
                            int K, int N, int Npad) {
  const int i = blockIdx.x * blockDim.x + threadIdx.x;
  if (i >= K * Npad) return;
  const int n = i / K, k = i - n * K;
  dst[n * K + k] = (n < N) ? (_Float16)src[k * N + n] : (_Float16)0.f;
}

// --------------- WMMA GEMM: C = [relu](A16[M,K] @ BT[N,K]^T + bias), async DMA
#define BM 64
#define BN 64
#define BK 32
__global__ void gemm_kernel(const _Float16* __restrict__ A16,
                            const _Float16* __restrict__ BT,   // [Npad][K]
                            const float* __restrict__ bias,
                            float* __restrict__ C,             // optional f32 out
                            _Float16* __restrict__ C16,        // optional f16 out
                            int M, int N, int K, int doRelu) {
  __shared__ __align__(16) _Float16 Ah[2][BM * BK];   // 2 x 4 KB [m][k]
  __shared__ __align__(16) _Float16 Bt[2][BN * BK];   // 2 x 4 KB [n][k]
  const int t = threadIdx.x;         // 0..255 (8 waves)
  const int lane = t & 31, wave = t >> 5;
  const int nbase = blockIdx.x * BN;
  const int mbase = blockIdx.y * BM;
  // wave -> two 16x16 tiles sharing one A fragment
  const int mi  = (wave * 2) >> 2;
  const int ni0 = (wave * 2) & 3, ni1 = (wave * 2 + 1) & 3;
  v8f acc0 = {}, acc1 = {};
  const int am = t >> 2, ak = (t & 3) << 3;   // both tiles: 64 rows x 32 halfs

  auto stage = [&](int k0, int buf) {
    async_ld_b128(&Ah[buf][am * BK + ak], A16,
                  ((unsigned)(mbase + am) * (unsigned)K + (unsigned)(k0 + ak)) * 2u);
    async_ld_b128(&Bt[buf][am * BK + ak], BT,
                  ((unsigned)(nbase + am) * (unsigned)K + (unsigned)(k0 + ak)) * 2u);
  };
  stage(0, 0);

  const int nk = K / BK;
  for (int kt = 0; kt < nk; ++kt) {
    const int buf = kt & 1;
    if (kt + 1 < nk) { stage((kt + 1) * BK, buf ^ 1); wait_async_le2(); }
    else             { wait_async_le0(); }
    __syncthreads();   // all waves' async data for this buffer visible
    const int arow = (mi << 4) + (lane & 15);
    const int ab   = ((lane >> 4) << 4);
    v16h a = *(const v16h*)&Ah[buf][arow * BK + ab];
    const int kb = ((lane >> 4) << 4);
    v16h b0 = *(const v16h*)&Bt[buf][((ni0 << 4) + (lane & 15)) * BK + kb];
    v16h b1 = *(const v16h*)&Bt[buf][((ni1 << 4) + (lane & 15)) * BK + kb];
    acc0 = wmma_f16f32(a, b0, acc0);
    acc1 = wmma_f16f32(a, b1, acc1);
    __syncthreads();   // buffer may be restaged next iteration
  }

  const int mrow = mbase + (mi << 4) + ((lane >> 4) << 3);
#pragma unroll
  for (int half = 0; half < 2; ++half) {
    const int ni = half ? ni1 : ni0;
    const v8f& acc = half ? acc1 : acc0;
    const int nn = nbase + (ni << 4) + (lane & 15);
    if (nn < N) {
      const float bv = bias[nn];
#pragma unroll
      for (int v = 0; v < 8; ++v) {
        float r = acc[v] + bv;
        if (doRelu) r = fmaxf(r, 0.f);
        if (C)   C[(mrow + v) * N + nn] = r;
        if (C16) C16[(mrow + v) * N + nn] = (_Float16)r;
      }
    }
  }
}

// ---------------------------------------------------------------- log-softmax
__global__ void logsoftmax_kernel(const float* __restrict__ logits, float* __restrict__ out) {
  const int n = blockIdx.x * blockDim.x + threadIdx.x;
  if (n >= N_PTS) return;
  float v[16];
  float m = -__builtin_inff();
#pragma unroll
  for (int i = 0; i < 16; ++i) { v[i] = logits[n * 16 + i]; m = fmaxf(m, v[i]); }
  float s = 0.f;
#pragma unroll
  for (int i = 0; i < 16; ++i) s += __expf(v[i] - m);
  const float lse = m + __logf(s);
#pragma unroll
  for (int i = 0; i < 16; ++i) out[n * 16 + i] = v[i] - lse;
}

// ================================================================ launcher
extern "C" void kernel_launch(void* const* d_in, const int* in_sizes, int n_in,
                              void* d_out, int out_size, void* d_ws, size_t ws_size,
                              hipStream_t stream) {
  (void)in_sizes; (void)n_in; (void)out_size; (void)ws_size;
  const float* x   = (const float*)d_in[0];
  const float* W1  = (const float*)d_in[1];
  const float* b1  = (const float*)d_in[2];
  const float* Wd1 = (const float*)d_in[3];
  const float* bd1 = (const float*)d_in[4];
  const float* Wd2 = (const float*)d_in[5];
  const float* bd2 = (const float*)d_in[6];
  const float* Wd3 = (const float*)d_in[7];
  const float* bd3 = (const float*)d_in[8];
  const float* Wl  = (const float*)d_in[9];
  const float* bl  = (const float*)d_in[10];
  const float* Wm1 = (const float*)d_in[11];
  const float* bm1 = (const float*)d_in[12];
  const float* Wm2 = (const float*)d_in[13];
  const float* bm2 = (const float*)d_in[14];
  const float* Wm3 = (const float*)d_in[15];
  const float* bm3 = (const float*)d_in[16];

  char* ws = (char*)d_ws;
  float*    xfeat = (float*)(ws + 0);            // 8192*64   f32
  float*    cat   = (float*)(ws + 2097152);      // 8192*192  f32 (x1|x2|x3)
  float*    logit = (float*)(ws + 8388608);      // 8192*16   f32
  _Float16* f16f  = (_Float16*)(ws + 8912896);   // 8192*64   f16
  float*    sq    = (float*)(ws + 9961472);      // 8192      f32
  int*      idx   = (int*)(ws + 9994240);        // 8192*20   i32
  _Float16* cat16 = (_Float16*)(ws + 10649600);  // 8192*192  f16
  _Float16* h1h   = (_Float16*)(ws + 13795328);  // 8192*1024 f16
  _Float16* h2h   = (_Float16*)(ws + 30572544);  // 8192*256  f16
  _Float16* h3h   = (_Float16*)(ws + 34766848);  // 8192*128  f16
  _Float16* WlT   = (_Float16*)(ws + 36864000);  // 1024*192  f16 (transposed)
  _Float16* Wm1T  = (_Float16*)(ws + 37257216);  // 256*1024  f16
  _Float16* Wm2T  = (_Float16*)(ws + 37781504);  // 128*256   f16
  _Float16* Wm3T  = (_Float16*)(ws + 37847040);  // 64*128    f16 (padded 16->64)

  // 1) kNN in 3-D + EdgeConv
  knn3_kernel<<<N_PTS / 128, 128, 0, stream>>>(x, idx);
  edgeconv_kernel<<<N_PTS, CH, 0, stream>>>(x, idx, W1, b1, xfeat);

  // 2) three dilate-flat layers (kNN in 64-D via async-DMA + WMMA)
  const float* fin = xfeat; int instride = CH;
  const float* Wds[3] = {Wd1, Wd2, Wd3};
  const float* bds[3] = {bd1, bd2, bd3};
  for (int l = 0; l < 3; ++l) {
    sqnorm_kernel<<<N_PTS, CH, 0, stream>>>(fin, instride, f16f, sq);
    knn64_kernel<<<N_PTS / KQT, KQT, 0, stream>>>(f16f, sq, idx);
    dilate_kernel<<<N_PTS, CH, 0, stream>>>(fin, instride, idx, Wds[l], bds[l],
                                            cat + l * CH, cat16 + l * CH, 192);
    fin = cat + l * CH; instride = 192;
  }

  // 3) transpose + f16-convert the head weights
  cvtT_kernel<<<(1024 * 192 + 255) / 256, 256, 0, stream>>>(Wl, WlT, 192, 1024, 1024);
  cvtT_kernel<<<(256 * 1024 + 255) / 256, 256, 0, stream>>>(Wm1, Wm1T, 1024, 256, 256);
  cvtT_kernel<<<(128 * 256 + 255) / 256, 256, 0, stream>>>(Wm2, Wm2T, 256, 128, 128);
  cvtT_kernel<<<(64 * 128 + 255) / 256, 256, 0, stream>>>(Wm3, Wm3T, 128, 16, 64);

  // 4) WMMA MLP head (f16 activations chained; f32 only for final logits)
  gemm_kernel<<<dim3(1024 / BN, N_PTS / BM), 256, 0, stream>>>(
      cat16, WlT, bl, nullptr, h1h, N_PTS, 1024, 192, 1);
  gemm_kernel<<<dim3(256 / BN, N_PTS / BM), 256, 0, stream>>>(
      h1h, Wm1T, bm1, nullptr, h2h, N_PTS, 256, 1024, 1);
  gemm_kernel<<<dim3(128 / BN, N_PTS / BM), 256, 0, stream>>>(
      h2h, Wm2T, bm2, nullptr, h3h, N_PTS, 128, 256, 1);
  gemm_kernel<<<dim3(1, N_PTS / BM), 256, 0, stream>>>(
      h3h, Wm3T, bm3, logit, nullptr, N_PTS, 16, 128, 0);

  // 5) log-softmax
  logsoftmax_kernel<<<N_PTS / 256, 256, 0, stream>>>(logit, (float*)d_out);
}
